// SwinTransformerBlock_73744588472692
// MI455X (gfx1250) — matrix operand order repaired and checked by reading, not compiled, not run on hardware
//
#include <hip/hip_runtime.h>

typedef __bf16 bf16_t;
typedef bf16_t v16bf __attribute__((ext_vector_type(16)));
typedef bf16_t v8bf  __attribute__((ext_vector_type(8)));
typedef float  v8f   __attribute__((ext_vector_type(8)));

#define WMMA_BF16(a, b, c) \
  __builtin_amdgcn_wmma_f32_16x16x32_bf16(false, (a), false, (b), (short)0, (c), false, false)

__device__ __forceinline__ bf16_t f2bf(float f) {
  unsigned u = __float_as_uint(f);
  u = u + 0x7FFFu + ((u >> 16) & 1u);
  unsigned short h = (unsigned short)(u >> 16);
  return __builtin_bit_cast(bf16_t, h);
}

__device__ __forceinline__ int rr_region(int x) { return x < 56 ? 0 : (x < 60 ? 1 : 2); }

// ---------------------------------------------------------------------------
// f32 -> bf16 weight conversion
// ---------------------------------------------------------------------------
__global__ void cvt_f32_bf16_k(const float* __restrict__ in, bf16_t* __restrict__ out, int n) {
  int i = blockIdx.x * 256 + threadIdx.x;
  if (i < n) out[i] = f2bf(in[i]);
}

// ---------------------------------------------------------------------------
// LN1 + cyclic shift(-4,-4) + window partition -> xw bf16 [32768,256]
// token t is window-major: t = (b*64 + win)*64 + tok
// ---------------------------------------------------------------------------
__global__ __launch_bounds__(256) void ln1_shift_window_k(
    const float* __restrict__ X, const float* __restrict__ g, const float* __restrict__ bt,
    bf16_t* __restrict__ xw) {
  const int t    = blockIdx.x * 8 + (threadIdx.x >> 5);
  const int lane = threadIdx.x & 31;
  const int wgi = t >> 6, tok = t & 63;
  const int b = wgi >> 6, w = wgi & 63;
  const int hs  = ((w >> 3) << 3) + (tok >> 3);
  const int wsc = ((w & 7)  << 3) + (tok & 7);
  const int hh = (hs + 4) & 63, wc = (wsc + 4) & 63;   // roll(-4) => read at +4
  const float* row = X + (((size_t)b * 4096) + hh * 64 + wc) * 256;

  float4 v0 = *(const float4*)(row + lane * 8);
  float4 v1 = *(const float4*)(row + lane * 8 + 4);
  float x[8] = {v0.x, v0.y, v0.z, v0.w, v1.x, v1.y, v1.z, v1.w};
  float s = 0.f, sq = 0.f;
#pragma unroll
  for (int e = 0; e < 8; ++e) { s += x[e]; sq += x[e] * x[e]; }
#pragma unroll
  for (int d = 1; d < 32; d <<= 1) { s += __shfl_xor(s, d, 32); sq += __shfl_xor(sq, d, 32); }
  float mu  = s * (1.f / 256.f);
  float var = sq * (1.f / 256.f) - mu * mu;
  float rs  = rsqrtf(var + 1e-5f);

  v8bf o;
#pragma unroll
  for (int e = 0; e < 8; ++e) {
    int c = lane * 8 + e;
    o[e] = f2bf((x[e] - mu) * rs * g[c] + bt[c]);
  }
  *(v8bf*)(xw + (size_t)t * 256 + lane * 8) = o;
}

// ---------------------------------------------------------------------------
// LN2 over hidden f32 [32768,256] (natural row order) -> y bf16
// ---------------------------------------------------------------------------
__global__ __launch_bounds__(256) void ln2_k(
    const float* __restrict__ X, const float* __restrict__ g, const float* __restrict__ bt,
    bf16_t* __restrict__ y) {
  const int t    = blockIdx.x * 8 + (threadIdx.x >> 5);
  const int lane = threadIdx.x & 31;
  const float* row = X + (size_t)t * 256;

  float4 v0 = *(const float4*)(row + lane * 8);
  float4 v1 = *(const float4*)(row + lane * 8 + 4);
  float x[8] = {v0.x, v0.y, v0.z, v0.w, v1.x, v1.y, v1.z, v1.w};
  float s = 0.f, sq = 0.f;
#pragma unroll
  for (int e = 0; e < 8; ++e) { s += x[e]; sq += x[e] * x[e]; }
#pragma unroll
  for (int d = 1; d < 32; d <<= 1) { s += __shfl_xor(s, d, 32); sq += __shfl_xor(sq, d, 32); }
  float mu  = s * (1.f / 256.f);
  float var = sq * (1.f / 256.f) - mu * mu;
  float rs  = rsqrtf(var + 1e-5f);

  v8bf o;
#pragma unroll
  for (int e = 0; e < 8; ++e) {
    int c = lane * 8 + e;
    o[e] = f2bf((x[e] - mu) * rs * g[c] + bt[c]);
  }
  *(v8bf*)(y + (size_t)t * 256 + lane * 8) = o;
}

// ---------------------------------------------------------------------------
// Tiled WMMA GEMM:  D[M,N] = A[M,K](bf16) @ W[N,K]^T(bf16) + bias, + epilogue
// Tile staging uses CDNA5 async global->LDS DMA (global_load_async_to_lds_b128,
// ASYNCcnt) with double-buffered LDS: tile k+32 is in flight while tile k is
// consumed by WMMAs.
// EPI 0: store bf16      EPI 1: exact GELU, store bf16
// EPI 2: window-reverse + unshift + residual(Res) -> f32 Out
// EPI 3: + Res -> f32 Out
// Block 256 thr (8 waves, 2x4), tile 128x128x32.
// ---------------------------------------------------------------------------
template <int EPI>
__global__ __launch_bounds__(256) void gemm_bf16_k(
    const bf16_t* __restrict__ A, const bf16_t* __restrict__ W,
    const float* __restrict__ bias, void* __restrict__ Out,
    const float* __restrict__ Res, int M, int N, int K) {
  __shared__ bf16_t As[2][128][48];
  __shared__ bf16_t Bs[2][128][48];

  const int tid = threadIdx.x;
  const int lane = tid & 31, wid = tid >> 5;
  const int lg = lane >> 4, lr = lane & 15;
  const int wm = wid >> 2, wn = wid & 3;
  const int mBase = blockIdx.x * 128;
  const int nBase = blockIdx.y * 128;

  // async DMA of one 128x32 bf16 tile pair into LDS buffer `buf`
  auto issue_tiles = [&](int buf, int k0) {
#pragma unroll
    for (int j = 0; j < 2; ++j) {
      int ci = tid + j * 256;            // 512 chunks of 16B per matrix
      int r = ci >> 2, c8 = (ci & 3) << 3;
      unsigned la = (unsigned)(uintptr_t)&As[buf][r][c8];
      unsigned lb = (unsigned)(uintptr_t)&Bs[buf][r][c8];
      const bf16_t* ga = A + (size_t)(mBase + r) * K + k0 + c8;
      const bf16_t* gb = W + (size_t)(nBase + r) * K + k0 + c8;
      asm volatile("global_load_async_to_lds_b128 %0, %1, off" :: "v"(la), "v"(ga) : "memory");
      asm volatile("global_load_async_to_lds_b128 %0, %1, off" :: "v"(lb), "v"(gb) : "memory");
    }
  };

  v8f acc[4][2] = {};

  issue_tiles(0, 0);
  asm volatile("s_wait_asynccnt 0x0" ::: "memory");
  __syncthreads();

  int cur = 0;
  for (int k0 = 0; k0 < K; k0 += 32) {
    if (k0 + 32 < K) issue_tiles(cur ^ 1, k0 + 32);  // prefetch next tile via async DMA

    v16bf af[4];
#pragma unroll
    for (int mt = 0; mt < 4; ++mt) {
      int row = wm * 64 + mt * 16 + lr;
      v8bf lo = *(const v8bf*)&As[cur][row][lg * 8];
      v8bf hi = *(const v8bf*)&As[cur][row][16 + lg * 8];
      af[mt] = __builtin_shufflevector(lo, hi, 0, 1, 2, 3, 4, 5, 6, 7, 8, 9, 10, 11, 12, 13, 14, 15);
    }
#pragma unroll
    for (int nt = 0; nt < 2; ++nt) {
      int col = wn * 32 + nt * 16 + lr;
      v8bf blo = *(const v8bf*)&Bs[cur][col][lg * 8];
      v8bf bhi = *(const v8bf*)&Bs[cur][col][16 + lg * 8];
      v16bf bfr = __builtin_shufflevector(blo, bhi, 0, 1, 2, 3, 4, 5, 6, 7, 8, 9, 10, 11, 12, 13, 14, 15);
#pragma unroll
      for (int mt = 0; mt < 4; ++mt) acc[mt][nt] = WMMA_BF16(af[mt], bfr, acc[mt][nt]);
    }

    asm volatile("s_wait_asynccnt 0x0" ::: "memory");
    __syncthreads();
    cur ^= 1;
  }

#pragma unroll
  for (int mt = 0; mt < 4; ++mt) {
#pragma unroll
    for (int nt = 0; nt < 2; ++nt) {
#pragma unroll
      for (int r = 0; r < 8; ++r) {
        int gm = mBase + wm * 64 + mt * 16 + lg * 8 + r;
        int gn = nBase + wn * 32 + nt * 16 + lr;
        float v = acc[mt][nt][r] + bias[gn];
        if constexpr (EPI == 0) {
          ((bf16_t*)Out)[(size_t)gm * N + gn] = f2bf(v);
        } else if constexpr (EPI == 1) {
          v = 0.5f * v * (1.0f + erff(v * 0.70710678118654752f));
          ((bf16_t*)Out)[(size_t)gm * N + gn] = f2bf(v);
        } else if constexpr (EPI == 2) {
          int wgi = gm >> 6, tok = gm & 63;
          int b = wgi >> 6, w = wgi & 63;
          int hs  = ((w >> 3) << 3) + (tok >> 3);
          int wsc = ((w & 7)  << 3) + (tok & 7);
          int hh = (hs + 4) & 63, wc = (wsc + 4) & 63;
          size_t o = (((size_t)b * 4096) + hh * 64 + wc) * 256 + gn;
          ((float*)Out)[o] = v + Res[o];
        } else {
          size_t o = (size_t)gm * 256 + gn;
          ((float*)Out)[o] = v + Res[o];
        }
      }
    }
  }
}

// ---------------------------------------------------------------------------
// Attention: one wave per (window, head). Block 256 = 8 waves = 8 heads.
// S = Q Kh^T * 1/sqrt(32) + relbias + swinmask; softmax; ctx = P Vh
// ---------------------------------------------------------------------------
__global__ __launch_bounds__(256) void attn_k(
    const bf16_t* __restrict__ Q, const bf16_t* __restrict__ Kt,
    const bf16_t* __restrict__ Vt, const float* __restrict__ relt,
    bf16_t* __restrict__ Ctx) {
  __shared__ bf16_t pst[8][16 * 64];

  const int wgi = blockIdx.x;            // 0..511  (b*64 + window)
  const int h = threadIdx.x >> 5;        // head = wave id
  const int lane = threadIdx.x & 31;
  const int lg = lane >> 4, lr = lane & 15;
  const int w = wgi & 63;
  const size_t base = (size_t)wgi * (64 * 256) + h * 32;

  // Q A-fragments (4 M-tiles, K=32 = full head dim)
  v16bf qf[4];
#pragma unroll
  for (int mt = 0; mt < 4; ++mt) {
    const bf16_t* p = Q + base + (size_t)(mt * 16 + lr) * 256;
    qf[mt] = *(const v16bf*)(p + lg * 16);
  }
  // K^T B-fragments (4 N-tiles over key tokens)
  v16bf kf[4];
#pragma unroll
  for (int nt = 0; nt < 4; ++nt) {
    const bf16_t* p = Kt + base + (size_t)(nt * 16 + lr) * 256;
    kf[nt] = *(const v16bf*)(p + lg * 16);
  }

  v8f sc[4][4] = {};
#pragma unroll
  for (int mt = 0; mt < 4; ++mt)
#pragma unroll
    for (int nt = 0; nt < 4; ++nt) sc[mt][nt] = WMMA_BF16(qf[mt], kf[nt], sc[mt][nt]);

  // Precompute key-side coords/regions (depend on nt, lr only)
  int jh[4], jw[4], regj[4];
#pragma unroll
  for (int nt = 0; nt < 4; ++nt) {
    int j = nt * 16 + lr;
    jh[nt] = j >> 3; jw[nt] = j & 7;
    int hs  = ((w >> 3) << 3) + jh[nt];
    int wsc = ((w & 7)  << 3) + jw[nt];
    regj[nt] = rr_region(hs) * 3 + rr_region(wsc);
  }

  const float scale = 0.17677669529663687f;  // 1/sqrt(32)
#pragma unroll
  for (int mt = 0; mt < 4; ++mt) {
#pragma unroll
    for (int r = 0; r < 8; ++r) {
      int i = mt * 16 + lg * 8 + r;      // query row (C-matrix layout)
      int ih = i >> 3, iw = i & 7;
      int hsI  = ((w >> 3) << 3) + ih;
      int wsI  = ((w & 7)  << 3) + iw;
      int regi = rr_region(hsI) * 3 + rr_region(wsI);
      float mx = -3.0e38f;
#pragma unroll
      for (int nt = 0; nt < 4; ++nt) {
        int idx = (ih - jh[nt] + 7) * 15 + (iw - jw[nt] + 7);
        float bias = relt[idx * 8 + h];
        float msk = (regi == regj[nt]) ? 0.f : -100.f;
        float v = sc[mt][nt][r] * scale + bias + msk;
        sc[mt][nt][r] = v;
        mx = fmaxf(mx, v);
      }
#pragma unroll
      for (int d = 1; d < 16; d <<= 1) mx = fmaxf(mx, __shfl_xor(mx, d, 32));
      float sum = 0.f;
#pragma unroll
      for (int nt = 0; nt < 4; ++nt) {
        float p = __expf(sc[mt][nt][r] - mx);
        sc[mt][nt][r] = p;
        sum += p;
      }
#pragma unroll
      for (int d = 1; d < 16; d <<= 1) sum += __shfl_xor(sum, d, 32);
      float inv = 1.0f / sum;
#pragma unroll
      for (int nt = 0; nt < 4; ++nt) sc[mt][nt][r] *= inv;
    }
  }

  // V B-fragments: [ks: token halves][nd: dim tiles]
  v16bf vf[2][2];
#pragma unroll
  for (int ks = 0; ks < 2; ++ks)
#pragma unroll
    for (int nd = 0; nd < 2; ++nd) {
      v16bf t;
#pragma unroll
      for (int e = 0; e < 16; ++e)
        t[e] = Vt[base + (size_t)(ks * 32 + (e < 8 ? lg * 8 + e : 8 + lg * 8 + e)) * 256 + nd * 16 + lr];
      vf[ks][nd] = t;
    }

  v8f cx[4][2] = {};
  bf16_t* ps = pst[h];
#pragma unroll
  for (int mt = 0; mt < 4; ++mt) {
    // stage P (C-layout) -> LDS, reread in A-layout (wave-private slice)
#pragma unroll
    for (int r = 0; r < 8; ++r)
#pragma unroll
      for (int nt = 0; nt < 4; ++nt)
        ps[(lg * 8 + r) * 64 + nt * 16 + lr] = f2bf(sc[mt][nt][r]);
    v16bf pf[2];
#pragma unroll
    for (int ks = 0; ks < 2; ++ks) {
      v8bf lo = *(const v8bf*)&ps[lr * 64 + ks * 32 + lg * 8];
      v8bf hi = *(const v8bf*)&ps[lr * 64 + ks * 32 + 16 + lg * 8];
      pf[ks] = __builtin_shufflevector(lo, hi, 0, 1, 2, 3, 4, 5, 6, 7, 8, 9, 10, 11, 12, 13, 14, 15);
    }
#pragma unroll
    for (int nd = 0; nd < 2; ++nd) {
      cx[mt][nd] = WMMA_BF16(pf[0], vf[0][nd], cx[mt][nd]);
      cx[mt][nd] = WMMA_BF16(pf[1], vf[1][nd], cx[mt][nd]);
    }
  }

  // store ctx in [token, h*32+d] layout (head-major channels)
#pragma unroll
  for (int mt = 0; mt < 4; ++mt)
#pragma unroll
    for (int nd = 0; nd < 2; ++nd)
#pragma unroll
      for (int r = 0; r < 8; ++r) {
        int row = mt * 16 + lg * 8 + r;
        Ctx[base + (size_t)row * 256 + nd * 16 + lr] = f2bf(cx[mt][nd][r]);
      }
}

// ---------------------------------------------------------------------------
extern "C" void kernel_launch(void* const* d_in, const int* in_sizes, int n_in,
                              void* d_out, int out_size, void* d_ws, size_t ws_size,
                              hipStream_t stream) {
  const float* hidden_states = (const float*)d_in[0];
  const float* ln1_g = (const float*)d_in[1];
  const float* ln1_b = (const float*)d_in[2];
  const float* wq = (const float*)d_in[3];
  const float* bq = (const float*)d_in[4];
  const float* wk = (const float*)d_in[5];
  const float* bk = (const float*)d_in[6];
  const float* wv = (const float*)d_in[7];
  const float* bv = (const float*)d_in[8];
  const float* rel_table = (const float*)d_in[9];
  const float* wo = (const float*)d_in[10];
  const float* bo = (const float*)d_in[11];
  const float* ln2_g = (const float*)d_in[12];
  const float* ln2_b = (const float*)d_in[13];
  const float* w1 = (const float*)d_in[14];
  const float* b1 = (const float*)d_in[15];
  const float* w2 = (const float*)d_in[16];
  const float* b2 = (const float*)d_in[17];

  char* ws = (char*)d_ws;
  const size_t TOKC_BF = (size_t)32768 * 256 * 2;  // 16 MiB

  bf16_t* xw  = (bf16_t*)(ws);                  // [0, 16M)
  bf16_t* qb  = (bf16_t*)(ws + 1 * TOKC_BF);    // [16M, 32M)
  bf16_t* kb  = (bf16_t*)(ws + 2 * TOKC_BF);    // [32M, 48M)
  bf16_t* vb  = (bf16_t*)(ws + 3 * TOKC_BF);    // [48M, 64M)
  bf16_t* ctx = xw;                             // alias (xw dead after QKV)
  bf16_t* h1  = (bf16_t*)(ws);                  // 64 MiB, aliases xw/q/k/v (all dead)
  float*  hid = (float*)(ws + 4 * TOKC_BF);     // 32 MiB f32
  bf16_t* yb  = (bf16_t*)(ws + 4 * TOKC_BF + (size_t)32768 * 256 * 4);
  char* wp = ws + 5 * TOKC_BF + (size_t)32768 * 256 * 4;
  bf16_t* wqb = (bf16_t*)wp;
  bf16_t* wkb = wqb + 65536;
  bf16_t* wvb = wkb + 65536;
  bf16_t* wob = wvb + 65536;
  bf16_t* w1b = wob + 65536;
  bf16_t* w2b = w1b + 262144;

  // 1) weight conversions
  cvt_f32_bf16_k<<<256, 256, 0, stream>>>(wq, wqb, 65536);
  cvt_f32_bf16_k<<<256, 256, 0, stream>>>(wk, wkb, 65536);
  cvt_f32_bf16_k<<<256, 256, 0, stream>>>(wv, wvb, 65536);
  cvt_f32_bf16_k<<<256, 256, 0, stream>>>(wo, wob, 65536);
  cvt_f32_bf16_k<<<1024, 256, 0, stream>>>(w1, w1b, 262144);
  cvt_f32_bf16_k<<<1024, 256, 0, stream>>>(w2, w2b, 262144);

  // 2) LN1 + shift + window partition
  ln1_shift_window_k<<<4096, 256, 0, stream>>>(hidden_states, ln1_g, ln1_b, xw);

  // 3) QKV projections
  dim3 g2(256, 2), g8(256, 8);
  gemm_bf16_k<0><<<g2, 256, 0, stream>>>(xw, wqb, bq, qb, nullptr, 32768, 256, 256);
  gemm_bf16_k<0><<<g2, 256, 0, stream>>>(xw, wkb, bk, kb, nullptr, 32768, 256, 256);
  gemm_bf16_k<0><<<g2, 256, 0, stream>>>(xw, wvb, bv, vb, nullptr, 32768, 256, 256);

  // 4) windowed attention (one wave per window-head)
  attn_k<<<512, 256, 0, stream>>>(qb, kb, vb, rel_table, ctx);

  // 5) output projection + window-reverse + unshift + residual -> hidden f32
  gemm_bf16_k<2><<<g2, 256, 0, stream>>>(ctx, wob, bo, hid, hidden_states, 32768, 256, 256);

  // 6) LN2
  ln2_k<<<4096, 256, 0, stream>>>(hid, ln2_g, ln2_b, yb);

  // 7) MLP
  gemm_bf16_k<1><<<g8, 256, 0, stream>>>(yb, w1b, b1, h1, nullptr, 32768, 1024, 256);
  gemm_bf16_k<3><<<g2, 256, 0, stream>>>(h1, w2b, b2, d_out, hid, 32768, 256, 1024);
}
